// WaveletMambaBlock_36558761624092
// MI455X (gfx1250) — compile-verified, matrix operand order checked by reading, not run on hardware
//
#include <hip/hip_runtime.h>
#include <hip/hip_bf16.h>

typedef __bf16 bf16;
typedef __attribute__((ext_vector_type(16))) __bf16 v16bf;
typedef __attribute__((ext_vector_type(8)))  __bf16 v8bf;
typedef __attribute__((ext_vector_type(8)))  float  v8f;

#define B_    8
#define DM    384
#define DI    768
#define DSN   16
#define C3_   1152
#define MLPH  1536
#define P1    8192    // B * 32*32  (mamba positions)
#define P2    32768   // B * 64*64  (full-res positions)

// ---------------------------------------------------------------------------
// workspace layout (bytes), all offsets 256B aligned; overlays verified by
// lifetime: m1 overlays [xll,xhigh,R1] (dead after fusion GEMM is consumed),
// m2 overlays [dt,ys], x2 overlays [hda,po].
// ---------------------------------------------------------------------------
static constexpr size_t o_wts   = 0;            // 7,225,344  bf16 weights
static constexpr size_t o_xll   = 7225344;      // 12,582,912 f32  (in-place ll_out)
static constexpr size_t o_xhigh = 19808256;     // 37,748,736 f32
static constexpr size_t o_R1    = 57556992;     // 50,331,648 f32  xz / fus
static constexpr size_t o_R2    = 107888640;    // 25,165,824      uf(f32)/recon(bf16)/xn(bf16)
static constexpr size_t o_R3    = 133054464;    // 12,582,912      ub(bf16)/ygated(bf16)
static constexpr size_t o_R4    = 145637376;    // 12,582,912      a1(bf16)/yo(f32)
static constexpr size_t o_dbl   = 158220288;    //  2,097,152 f32
static constexpr size_t o_dtrp  = 160317440;    //    524,288 bf16
static constexpr size_t o_dt    = 160841728;    // 25,165,824 f32  (m2 overlays dt+ys)
static constexpr size_t o_ys    = 186007552;    // 25,165,824 f32
static constexpr size_t o_hda   = 211173376;    // 18,874,368 bf16 (x2 overlays hda+po)
static constexpr size_t o_po    = 230047744;    // 37,748,736 f32
static constexpr size_t o_m1    = o_xll;        // 100,663,296 bf16
static constexpr size_t o_m2    = o_dt;         //  50,331,648 f32
static constexpr size_t o_x2    = o_hda;        //  50,331,648 f32

// weight sub-offsets (bf16 elements from o_wts)
static constexpr size_t w_in  = 0;        // 1536x384
static constexpr size_t w_x   = 589824;   // 64x768 (rows>=56 zero)
static constexpr size_t w_dt  = 638976;   // 768x32 (cols>=24 zero)
static constexpr size_t w_out = 663552;   // 384x768
static constexpr size_t w_pw  = 958464;   // 1152x1152
static constexpr size_t w_fus = 2285568;  // 384x384
static constexpr size_t w_m1  = 2433024;  // 1536x384
static constexpr size_t w_m2  = 3022848;  // 384x1536

__device__ __forceinline__ float siluf(float x) { return x / (1.f + __expf(-x)); }

__device__ __forceinline__ v16bf cat8(v8bf lo, v8bf hi) {
  v16bf r;
#pragma unroll
  for (int t = 0; t < 8; ++t) { r[t] = lo[t]; r[t + 8] = hi[t]; }
  return r;
}

// ---------------------------------------------------------------------------
// WMMA GEMM: C[M,N] = A[M,K] * W[N,K]^T , bf16 inputs, f32 accumulate.
// block = 128 threads (4 waves), block tile 64x64, wave tile 32x32 (2x2 WMMA).
// EPI 0: store f32.  EPI 1: store bf16 of gelu(x + bias).
// M%64==0, N%64==0, K%32==0 (guaranteed by the padded buffers).
// ---------------------------------------------------------------------------
template <int EPI>
__device__ __forceinline__ void epi_store(v8f acc, int m0, int n, int N,
                                          float* Cf, bf16* Cb, const float* bias) {
#pragma unroll
  for (int r = 0; r < 8; ++r) {
    float v = acc[r];
    size_t idx = (size_t)(m0 + r) * N + n;
    if (EPI == 0) {
      Cf[idx] = v;
    } else {
      v += bias[n];
      v = 0.5f * v * (1.0f + erff(v * 0.70710678118f));
      Cb[idx] = (bf16)v;
    }
  }
}

template <int EPI>
__global__ __launch_bounds__(128) void wmma_gemm(
    const bf16* __restrict__ A, const bf16* __restrict__ W,
    float* __restrict__ Cf, bf16* __restrict__ Cb,
    const float* __restrict__ bias, int M, int N, int K) {
  const int wave = threadIdx.x >> 5;
  const int lane = threadIdx.x & 31;
  const int half = lane >> 4;
  const int l16  = lane & 15;
  const int mBase = blockIdx.y * 64 + (wave >> 1) * 32;
  const int nBase = blockIdx.x * 64 + (wave & 1) * 32;
  const int koff = half * 8;

  const bf16* a0p = A + (size_t)(mBase + l16) * K + koff;
  const bf16* a1p = A + (size_t)(mBase + 16 + l16) * K + koff;
  const bf16* b0p = W + (size_t)(nBase + l16) * K + koff;
  const bf16* b1p = W + (size_t)(nBase + 16 + l16) * K + koff;

  v8f acc00 = {}, acc01 = {}, acc10 = {}, acc11 = {};

  for (int kk = 0; kk < K; kk += 32) {
    v16bf a0 = cat8(*(const v8bf*)(a0p + kk), *(const v8bf*)(a0p + kk + 16));
    v16bf a1 = cat8(*(const v8bf*)(a1p + kk), *(const v8bf*)(a1p + kk + 16));
    v16bf b0 = cat8(*(const v8bf*)(b0p + kk), *(const v8bf*)(b0p + kk + 16));
    v16bf b1 = cat8(*(const v8bf*)(b1p + kk), *(const v8bf*)(b1p + kk + 16));
    if (kk + 32 < K) {               // gfx1250: lowers to global_prefetch_b8
      __builtin_prefetch(a0p + kk + 32, 0, 3);
      __builtin_prefetch(b0p + kk + 32, 0, 3);
    }
    acc00 = __builtin_amdgcn_wmma_f32_16x16x32_bf16(false, a0, false, b0, (short)0, acc00, false, false);
    acc01 = __builtin_amdgcn_wmma_f32_16x16x32_bf16(false, a0, false, b1, (short)0, acc01, false, false);
    acc10 = __builtin_amdgcn_wmma_f32_16x16x32_bf16(false, a1, false, b0, (short)0, acc10, false, false);
    acc11 = __builtin_amdgcn_wmma_f32_16x16x32_bf16(false, a1, false, b1, (short)0, acc11, false, false);
  }

  const int n0 = nBase + l16, n1 = nBase + 16 + l16;
  const int m0 = mBase + half * 8, m1 = mBase + 16 + half * 8;
  epi_store<EPI>(acc00, m0, n0, N, Cf, Cb, bias);
  epi_store<EPI>(acc01, m0, n1, N, Cf, Cb, bias);
  epi_store<EPI>(acc10, m1, n0, N, Cf, Cb, bias);
  epi_store<EPI>(acc11, m1, n1, N, Cf, Cb, bias);
}

// ---------------------------------------------------------------------------
// f32 -> bf16 weight convert with optional zero padding (and strided source).
// ---------------------------------------------------------------------------
__global__ void k_cvt(const float* __restrict__ src, bf16* __restrict__ dst,
                      int Ndst, int Kdst, int Nsrc, int Kvalid, int srcStride) {
  int i = blockIdx.x * 256 + threadIdx.x;
  if (i >= Ndst * Kdst) return;
  int n = i / Kdst, k = i % Kdst;
  float v = (n < Nsrc && k < Kvalid) ? src[(size_t)n * srcStride + k] : 0.f;
  dst[i] = (bf16)v;
}

// ---------------------------------------------------------------------------
// DWT: x (B,384,64,64) -> ll (B,384,32,32), high (B,1152,32,32)
// ---------------------------------------------------------------------------
__global__ void k_dwt(const float* __restrict__ x, float* __restrict__ xll,
                      float* __restrict__ xh) {
  int i = blockIdx.x * 256 + threadIdx.x;
  if (i >= B_ * DM * 1024) return;
  int w = i & 31, h = (i >> 5) & 31, c = (i >> 10) % DM, b = i / (DM * 1024);
  const float* xb = x + ((size_t)(b * DM + c)) * 4096;
  float x1 = xb[(2 * h) * 64 + 2 * w] * 0.5f;
  float x3 = xb[(2 * h) * 64 + 2 * w + 1] * 0.5f;
  float x2 = xb[(2 * h + 1) * 64 + 2 * w] * 0.5f;
  float x4 = xb[(2 * h + 1) * 64 + 2 * w + 1] * 0.5f;
  int hw = h * 32 + w;
  xll[((size_t)(b * DM + c)) * 1024 + hw] = x1 + x2 + x3 + x4;
  xh[((size_t)(b * C3_ + 0 * DM + c)) * 1024 + hw] = -x1 - x2 + x3 + x4;  // HL
  xh[((size_t)(b * C3_ + 1 * DM + c)) * 1024 + hw] = -x1 + x2 - x3 + x4;  // LH
  xh[((size_t)(b * C3_ + 2 * DM + c)) * 1024 + hw] =  x1 - x2 - x3 + x4;  // HH
}

// ---------------------------------------------------------------------------
// LayerNorm over channels of NCHW fp32 -> pos-major bf16 (one wave32 / pos)
// ---------------------------------------------------------------------------
__global__ void k_ln(const float* __restrict__ X, const float* __restrict__ g,
                     const float* __restrict__ bb, bf16* __restrict__ out,
                     int C, int HW, int P) {
  int gw = (blockIdx.x * blockDim.x + threadIdx.x) >> 5;
  int lane = threadIdx.x & 31;
  if (gw >= P) return;
  int b = gw / HW, hw = gw % HW;
  const float* base = X + ((size_t)b * C) * HW + hw;
  float s = 0.f, s2 = 0.f;
  for (int c = lane; c < C; c += 32) {
    float v = base[(size_t)c * HW];
    s += v; s2 += v * v;
  }
#pragma unroll
  for (int m = 16; m > 0; m >>= 1) {
    s  += __shfl_xor(s, m, 32);
    s2 += __shfl_xor(s2, m, 32);
  }
  float mean = s / C;
  float rstd = rsqrtf(s2 / C - mean * mean + 1e-5f);
  bf16* op = out + (size_t)gw * C;
  for (int c = lane; c < C; c += 32) {
    float v = base[(size_t)c * HW];
    op[c] = (bf16)((v - mean) * rstd * g[c] + bb[c]);
  }
}

// causal depthwise conv1d(k=4) + bias + silu over xi = xz[:, :768]
__global__ void k_conv(const float* __restrict__ xz, const float* __restrict__ cw,
                       const float* __restrict__ cb, float* __restrict__ uf,
                       bf16* __restrict__ ub) {
  int i = blockIdx.x * 256 + threadIdx.x;
  if (i >= P1 * DI) return;
  int d = i % DI, p = i / DI, l = p & 1023;
  float acc = cb[d];
#pragma unroll
  for (int j = 0; j < 4; ++j) {
    int l2 = l - 3 + j;
    if (l2 >= 0) acc += cw[d * 4 + j] * xz[(size_t)(p - 3 + j) * 1536 + d];
  }
  float s = siluf(acc);
  uf[i] = s;
  ub[i] = (bf16)s;
}

// dtraw += dtproj_b; softplus (in place)
__global__ void k_softplus(float* __restrict__ dt, const float* __restrict__ dtb) {
  int i = blockIdx.x * 256 + threadIdx.x;
  if (i >= P1 * DI) return;
  float v = dt[i] + dtb[i % DI];
  dt[i] = (v > 20.f) ? v : log1pf(__expf(v));
}

// selective scan: thread = (b,d) channel, 16 states in regs, B/C staged in LDS
__global__ __launch_bounds__(256) void k_scan(const float* __restrict__ dt,
                                              const float* __restrict__ dbl,
                                              const float* __restrict__ uf,
                                              const float* __restrict__ A_log,
                                              float* __restrict__ ys) {
  int b = blockIdx.x / 3;
  int d = (blockIdx.x % 3) * 256 + threadIdx.x;
  float A[DSN], h[DSN];
#pragma unroll
  for (int n = 0; n < DSN; ++n) { A[n] = -__expf(A_log[d * DSN + n]); h[n] = 0.f; }
  __shared__ float sB[DSN], sC[DSN];
  for (int l = 0; l < 1024; ++l) {
    int p = b * 1024 + l;
    if (threadIdx.x < DSN) {
      sB[threadIdx.x] = dbl[(size_t)p * 64 + 24 + threadIdx.x];
      sC[threadIdx.x] = dbl[(size_t)p * 64 + 40 + threadIdx.x];
    }
    __syncthreads();
    float dtv = dt[(size_t)p * DI + d];
    float du = dtv * uf[(size_t)p * DI + d];
    float y = 0.f;
#pragma unroll
    for (int n = 0; n < DSN; ++n) {
      h[n] = __expf(dtv * A[n]) * h[n] + du * sB[n];
      y += h[n] * sC[n];
    }
    ys[(size_t)p * DI + d] = y;
    __syncthreads();
  }
}

// y = (ys + u*Dskip) * silu(z) -> bf16
__global__ void k_gate(const float* __restrict__ ys, const float* __restrict__ uf,
                       const float* __restrict__ xz, const float* __restrict__ Dskip,
                       bf16* __restrict__ yg) {
  int i = blockIdx.x * 256 + threadIdx.x;
  if (i >= P1 * DI) return;
  int d = i % DI, p = i / DI;
  float y = ys[i] + uf[i] * Dskip[d];
  float z = xz[(size_t)p * 1536 + 768 + d];
  yg[i] = (bf16)(y * siluf(z));
}

// x_ll += y^T (pos-major -> NCHW), in place
__global__ void k_addll(float* __restrict__ xll, const float* __restrict__ yo) {
  int i = blockIdx.x * 256 + threadIdx.x;
  if (i >= P1 * DM) return;
  int c = i % DM, p = i / DM, b = p >> 10, hw = p & 1023;
  xll[((size_t)(b * DM + c)) * 1024 + hw] += yo[i];
}

// depthwise 3x3 (pad 1) + BN + silu -> pos-major bf16 rows for pw GEMM
__global__ void k_dwconv(const float* __restrict__ xh, const float* __restrict__ dww,
                         const float* __restrict__ bng, const float* __restrict__ bnb,
                         const float* __restrict__ bnm, const float* __restrict__ bnv,
                         bf16* __restrict__ hda) {
  int i = blockIdx.x * 256 + threadIdx.x;
  if (i >= B_ * C3_ * 1024) return;
  int w = i & 31, h = (i >> 5) & 31, c = (i >> 10) % C3_, b = i / (C3_ * 1024);
  const float* xb = xh + ((size_t)(b * C3_ + c)) * 1024;
  float acc = 0.f;
#pragma unroll
  for (int dy = 0; dy < 3; ++dy) {
    int hy = h + dy - 1;
    if (hy < 0 || hy > 31) continue;
#pragma unroll
    for (int dx = 0; dx < 3; ++dx) {
      int wx = w + dx - 1;
      if (wx < 0 || wx > 31) continue;
      acc += dww[c * 9 + dy * 3 + dx] * xb[hy * 32 + wx];
    }
  }
  float v = (acc - bnm[c]) * rsqrtf(bnv[c] + 1e-5f) * bng[c] + bnb[c];
  hda[((size_t)(b * 1024 + h * 32 + w)) * C3_ + c] = (bf16)siluf(v);
}

// IDWT of (ll_out, high + po^T) -> recon NHWC bf16
__global__ void k_idwt(const float* __restrict__ xll, const float* __restrict__ xh,
                       const float* __restrict__ po, bf16* __restrict__ recon) {
  int i = blockIdx.x * 256 + threadIdx.x;
  if (i >= B_ * DM * 1024) return;
  int w = i & 31, h = (i >> 5) & 31, c = (i >> 10) % DM, b = i / (DM * 1024);
  int hw = h * 32 + w;
  size_t ppos = (size_t)(b * 1024 + hw) * C3_;
  float ll = xll[((size_t)(b * DM + c)) * 1024 + hw] * 0.5f;
  float hl = (xh[((size_t)(b * C3_ + 0 * DM + c)) * 1024 + hw] + po[ppos + 0 * DM + c]) * 0.5f;
  float lh = (xh[((size_t)(b * C3_ + 1 * DM + c)) * 1024 + hw] + po[ppos + 1 * DM + c]) * 0.5f;
  float hh = (xh[((size_t)(b * C3_ + 2 * DM + c)) * 1024 + hw] + po[ppos + 2 * DM + c]) * 0.5f;
  float y1 = ll - hl - lh + hh;
  float y2 = ll - hl + lh - hh;
  float y3 = ll + hl - lh - hh;
  float y4 = ll + hl + lh + hh;
  size_t base = ((size_t)b * 64 + 2 * h) * 64;
  recon[(base + 2 * w) * DM + c]            = (bf16)y1;
  recon[(base + 2 * w + 1) * DM + c]        = (bf16)y3;
  recon[(base + 64 + 2 * w) * DM + c]       = (bf16)y2;
  recon[(base + 64 + 2 * w + 1) * DM + c]   = (bf16)y4;
}

// x2 = shortcut + fus^T + fus_b  (NCHW f32)
__global__ void k_x2(const float* __restrict__ fus, const float* __restrict__ fb,
                     const float* __restrict__ x, float* __restrict__ x2) {
  int i = blockIdx.x * 256 + threadIdx.x;
  if (i >= P2 * DM) return;
  int c = i % DM, p = i / DM, b = p >> 12, yx = p & 4095;
  size_t idx = ((size_t)(b * DM + c)) * 4096 + yx;
  x2[idx] = fus[i] + fb[c] + x[idx];
}

// out = x2 + (m2 + b2)^T  (NCHW f32)
__global__ void k_final(const float* __restrict__ x2, const float* __restrict__ m2,
                        const float* __restrict__ b2, float* __restrict__ out) {
  int i = blockIdx.x * 256 + threadIdx.x;
  if (i >= P2 * DM) return;
  int c = i % DM, p = i / DM, b = p >> 12, yx = p & 4095;
  size_t idx = ((size_t)(b * DM + c)) * 4096 + yx;
  out[idx] = x2[idx] + m2[i] + b2[c];
}

// ---------------------------------------------------------------------------
extern "C" void kernel_launch(void* const* d_in, const int* in_sizes, int n_in,
                              void* d_out, int out_size, void* d_ws, size_t ws_size,
                              hipStream_t stream) {
  const float* x      = (const float*)d_in[0];
  const float* ln1_g  = (const float*)d_in[1];
  const float* ln1_b  = (const float*)d_in[2];
  const float* in_w   = (const float*)d_in[3];
  const float* conv_w = (const float*)d_in[4];
  const float* conv_b = (const float*)d_in[5];
  const float* xproj_w  = (const float*)d_in[6];
  const float* dtproj_w = (const float*)d_in[7];
  const float* dtproj_b = (const float*)d_in[8];
  const float* A_log  = (const float*)d_in[9];
  const float* Dskip  = (const float*)d_in[10];
  const float* out_w  = (const float*)d_in[11];
  const float* dw_w   = (const float*)d_in[12];
  const float* bn_g   = (const float*)d_in[13];
  const float* bn_b   = (const float*)d_in[14];
  const float* bn_m   = (const float*)d_in[15];
  const float* bn_v   = (const float*)d_in[16];
  const float* pw_w   = (const float*)d_in[17];
  const float* fus_w  = (const float*)d_in[18];
  const float* fus_b  = (const float*)d_in[19];
  const float* ln2_g  = (const float*)d_in[20];
  const float* ln2_b  = (const float*)d_in[21];
  const float* mlp_w1 = (const float*)d_in[22];
  const float* mlp_b1 = (const float*)d_in[23];
  const float* mlp_w2 = (const float*)d_in[24];
  const float* mlp_b2 = (const float*)d_in[25];
  float* outp = (float*)d_out;

  char* ws = (char*)d_ws;
  bf16*  wts   = (bf16*)(ws + o_wts);
  float* xll   = (float*)(ws + o_xll);
  float* xhigh = (float*)(ws + o_xhigh);
  float* xz    = (float*)(ws + o_R1);   // then fus
  float* fus   = (float*)(ws + o_R1);
  float* uf    = (float*)(ws + o_R2);
  bf16*  recon = (bf16*)(ws + o_R2);
  bf16*  xn    = (bf16*)(ws + o_R2);
  bf16*  ub    = (bf16*)(ws + o_R3);
  bf16*  yg    = (bf16*)(ws + o_R3);
  bf16*  a1    = (bf16*)(ws + o_R4);
  float* yo    = (float*)(ws + o_R4);
  float* dbl   = (float*)(ws + o_dbl);
  bf16*  dtrp  = (bf16*)(ws + o_dtrp);
  float* dt    = (float*)(ws + o_dt);
  float* ys    = (float*)(ws + o_ys);
  bf16*  hda   = (bf16*)(ws + o_hda);
  float* po    = (float*)(ws + o_po);
  bf16*  m1    = (bf16*)(ws + o_m1);
  float* m2    = (float*)(ws + o_m2);
  float* x2    = (float*)(ws + o_x2);

  auto cdiv = [](long a, long b) { return (int)((a + b - 1) / b); };
  dim3 blk1(256), blkG(128);

  // --- weight -> bf16 (with padding where needed) ---
  k_cvt<<<cdiv(1536L * 384, 256), blk1, 0, stream>>>(in_w,     wts + w_in,  1536, 384, 1536, 384, 384);
  k_cvt<<<cdiv(64L * 768,   256), blk1, 0, stream>>>(xproj_w,  wts + w_x,   64,   768, 56,   768, 768);
  k_cvt<<<cdiv(768L * 32,   256), blk1, 0, stream>>>(dtproj_w, wts + w_dt,  768,  32,  768,  24,  24);
  k_cvt<<<cdiv(384L * 768,  256), blk1, 0, stream>>>(out_w,    wts + w_out, 384,  768, 384,  768, 768);
  k_cvt<<<cdiv(1152L * 1152,256), blk1, 0, stream>>>(pw_w,     wts + w_pw,  1152, 1152,1152, 1152,1152);
  k_cvt<<<cdiv(384L * 384,  256), blk1, 0, stream>>>(fus_w,    wts + w_fus, 384,  384, 384,  384, 384);
  k_cvt<<<cdiv(1536L * 384, 256), blk1, 0, stream>>>(mlp_w1,   wts + w_m1,  1536, 384, 1536, 384, 384);
  k_cvt<<<cdiv(384L * 1536, 256), blk1, 0, stream>>>(mlp_w2,   wts + w_m2,  384,  1536,384,  1536,1536);

  // --- DWT ---
  k_dwt<<<cdiv(B_ * DM * 1024L, 256), blk1, 0, stream>>>(x, xll, xhigh);

  // --- LN1 -> a1 (bf16, pos-major) ---
  k_ln<<<P1 / 8, blk1, 0, stream>>>(xll, ln1_g, ln1_b, a1, DM, 1024, P1);

  // --- in_proj: xz = a1 @ in_w^T  (8192x1536, K=384) ---
  wmma_gemm<0><<<dim3(1536 / 64, P1 / 64), blkG, 0, stream>>>(a1, wts + w_in, xz, nullptr, nullptr, P1, 1536, 384);

  // --- causal dwconv1d + silu -> uf/ub ---
  k_conv<<<cdiv(P1 * (long)DI, 256), blk1, 0, stream>>>(xz, conv_w, conv_b, uf, ub);

  // --- x_proj: dbl = ub @ xw^T (8192x64, K=768) ---
  wmma_gemm<0><<<dim3(64 / 64, P1 / 64), blkG, 0, stream>>>(ub, wts + w_x, dbl, nullptr, nullptr, P1, 64, 768);

  // --- pack dtr (cols 0..23 of dbl) into padded bf16 8192x32 ---
  k_cvt<<<cdiv(P1 * 32L, 256), blk1, 0, stream>>>(dbl, dtrp, P1, 32, P1, 24, 64);

  // --- dt_proj: dt = dtrp @ dtw^T (8192x768, K=32), then +bias, softplus ---
  wmma_gemm<0><<<dim3(768 / 64, P1 / 64), blkG, 0, stream>>>(dtrp, wts + w_dt, dt, nullptr, nullptr, P1, 768, 32);
  k_softplus<<<cdiv(P1 * (long)DI, 256), blk1, 0, stream>>>(dt, dtproj_b);

  // --- selective scan ---
  k_scan<<<B_ * 3, blk1, 0, stream>>>(dt, dbl, uf, A_log, ys);

  // --- gate: yg = (ys + u*Dskip) * silu(z) ---
  k_gate<<<cdiv(P1 * (long)DI, 256), blk1, 0, stream>>>(ys, uf, xz, Dskip, yg);

  // --- out_proj: yo = yg @ out_w^T (8192x384, K=768); xll += yo^T ---
  wmma_gemm<0><<<dim3(384 / 64, P1 / 64), blkG, 0, stream>>>(yg, wts + w_out, yo, nullptr, nullptr, P1, 384, 768);
  k_addll<<<cdiv(P1 * (long)DM, 256), blk1, 0, stream>>>(xll, yo);

  // --- high band: dw3x3+BN+silu -> hda; pointwise GEMM 1152x1152 ---
  k_dwconv<<<cdiv(B_ * C3_ * 1024L, 256), blk1, 0, stream>>>(xhigh, dw_w, bn_g, bn_b, bn_m, bn_v, hda);
  wmma_gemm<0><<<dim3(1152 / 64, P1 / 64), blkG, 0, stream>>>(hda, wts + w_pw, po, nullptr, nullptr, P1, 1152, 1152);

  // --- IDWT -> recon (NHWC bf16); fusion GEMM 384x384; x2 = shortcut + . ---
  k_idwt<<<cdiv(B_ * DM * 1024L, 256), blk1, 0, stream>>>(xll, xhigh, po, recon);
  wmma_gemm<0><<<dim3(384 / 64, P2 / 64), blkG, 0, stream>>>(recon, wts + w_fus, fus, nullptr, nullptr, P2, 384, 384);
  k_x2<<<cdiv(P2 * (long)DM, 256), blk1, 0, stream>>>(fus, fus_b, x, x2);

  // --- LN2 -> xn; MLP (fused bias+gelu epilogue on mlp1) ---
  k_ln<<<P2 / 8, blk1, 0, stream>>>(x2, ln2_g, ln2_b, xn, DM, 4096, P2);
  wmma_gemm<1><<<dim3(1536 / 64, P2 / 64), blkG, 0, stream>>>(xn, wts + w_m1, nullptr, m1, mlp_b1, P2, 1536, 384);
  wmma_gemm<0><<<dim3(384 / 64, P2 / 64), blkG, 0, stream>>>(m1, wts + w_m2, m2, nullptr, nullptr, P2, 384, 1536);

  // --- out = x2 + (m2 + b2)^T ---
  k_final<<<cdiv(P2 * (long)DM, 256), blk1, 0, stream>>>(x2, m2, mlp_b2, outp);

  (void)in_sizes; (void)n_in; (void)out_size; (void)ws_size;
}
